// HyperbolicHead_9397388443805
// MI455X (gfx1250) — matrix operand order repaired and checked by reading, not compiled, not run on hardware
//
#include <hip/hip_runtime.h>

// Sizes from the reference: B=8, N=4096, D=512. Output [1,B,N,N] fp32.
#define BB 8
#define NN 4096
#define DD 512

typedef __attribute__((ext_vector_type(16))) __bf16 v16bf;
typedef __attribute__((ext_vector_type(8)))  __bf16 v8bf;
typedef __attribute__((ext_vector_type(8)))  float  v8f;
typedef __attribute__((ext_vector_type(4)))  float  f4;

__device__ __forceinline__ f4 ld4(const float* p) { return *(const f4*)p; }
__device__ __forceinline__ v8bf ld8bf(const __bf16* p) { return *(const v8bf*)p; }
__device__ __forceinline__ v16bf cat16(v8bf a, v8bf b) {
    return __builtin_shufflevector(a, b, 0, 1, 2, 3, 4, 5, 6, 7,
                                         8, 9, 10, 11, 12, 13, 14, 15);
}

// ---------------------------------------------------------------------------
// Prep kernel: one pass over X producing
//   sq[row]  = ||x_row||^2
//   xhi[row] = bf16(x), xlo[row] = bf16(x - f32(hi))   (row-major [B*N, D])
// ---------------------------------------------------------------------------
__global__ __launch_bounds__(256)
void hyp_prep_kernel(const float* __restrict__ x, float* __restrict__ sq,
                     __bf16* __restrict__ xhi, __bf16* __restrict__ xlo) {
    const int row  = blockIdx.x * 8 + (threadIdx.x >> 5);
    const int lane = threadIdx.x & 31;
    const float* xr = x + (size_t)row * DD;
    __bf16* hr = xhi + (size_t)row * DD;
    __bf16* lr = xlo + (size_t)row * DD;
    float s = 0.0f;
#pragma unroll
    for (int i = 0; i < DD / 32; ++i) {
        const int idx = lane + 32 * i;
        const float v = xr[idx];
        const __bf16 hi = (__bf16)v;
        hr[idx] = hi;
        lr[idx] = (__bf16)(v - (float)hi);
        s = fmaf(v, v, s);
    }
#pragma unroll
    for (int m = 16; m >= 1; m >>= 1) s += __shfl_xor(s, m, 32);
    if (lane == 0) sq[row] = s;
}

// Plain row-norm kernel (fallback path when scratch can't hold xhi/xlo).
__global__ __launch_bounds__(256)
void hyp_rownorm_kernel(const float* __restrict__ x, float* __restrict__ sq) {
    const int row  = blockIdx.x * 8 + (threadIdx.x >> 5);
    const int lane = threadIdx.x & 31;
    const float* xr = x + (size_t)row * DD;
    float s = 0.0f;
#pragma unroll
    for (int i = 0; i < DD / 32; ++i) {
        const float v = xr[lane + 32 * i];
        s = fmaf(v, v, s);
    }
#pragma unroll
    for (int m = 16; m >= 1; m >>= 1) s += __shfl_xor(s, m, 32);
    if (lane == 0) sq[row] = s;
}

// ---------------------------------------------------------------------------
// Shared epilogue for one 16x64 strip of accumulators.
// acc layout per ISA: VGPR r, lane half h -> row r + 8h, col lane&15.
// ---------------------------------------------------------------------------
__device__ __forceinline__ void hyp_epilogue(const v8f acc[4],
                                             const float* __restrict__ sqb,
                                             float* __restrict__ out,
                                             int b, int rowBase, int colBase,
                                             int h, int lm) {
    float sqi[8];
    {
        const f4 s0 = ld4(sqb + rowBase + h * 8);
        const f4 s1 = ld4(sqb + rowBase + h * 8 + 4);
#pragma unroll
        for (int e = 0; e < 4; ++e) { sqi[e] = s0[e]; sqi[4 + e] = s1[e]; }
    }
#pragma unroll
    for (int j = 0; j < 4; ++j) {
        const int col   = colBase + j * 16 + lm;
        const float sqj = sqb[col];
        const float omj = 1.0f - sqj;
#pragma unroll
        for (int r = 0; r < 8; ++r) {
            const int row  = rowBase + h * 8 + r;
            const float g  = acc[j][r];
            float d2 = fmaxf(sqi[r] + sqj - 2.0f * g, 0.0f);
            const float dn  = (d2 > 0.0f) ? sqrtf(d2) : 0.0f;
            const float arg = fmaf(2.0f, dn / ((1.0f - sqi[r]) * omj), 1.0f);
            const float dist =
                (arg > 1.0f) ? logf(arg + sqrtf(fmaf(arg, arg, -1.0f))) : 0.0f;
            out[((size_t)(b * NN + row)) * NN + col] = dist;
        }
    }
}

// ---------------------------------------------------------------------------
// Fast GEMM: 32(M) x 64(N) strip per wave, pre-split bf16 hi/lo inputs.
// Per K-chunk (K=32): 24 global_load_b128 and 24 v_wmma_f32_16x16x32_bf16
// (each B fragment feeds 6 WMMAs across 2 M-tiles). All inner-loop addresses
// are 4 incremented base pointers + constant immediate offsets.
// Block = 128 threads = 4 waves; block covers 128x64. Grid (N/64, N/128, B).
// ---------------------------------------------------------------------------
__global__ __launch_bounds__(128)
void hyp_dist_bf16_kernel(const __bf16* __restrict__ xhi,
                          const __bf16* __restrict__ xlo,
                          const float* __restrict__ sq,
                          float* __restrict__ out) {
    const int lane    = threadIdx.x & 31;
    const int wave    = threadIdx.x >> 5;
    const int b       = blockIdx.z;
    const int rowBase = (blockIdx.y * 4 + wave) * 32;   // 32-row strip
    const int colBase = blockIdx.x * 64;                // 64-col strip
    const int h       = lane >> 4;
    const int lm      = lane & 15;

    const __bf16* Hb = xhi + (size_t)b * NN * DD;
    const __bf16* Lb = xlo + (size_t)b * NN * DD;

    // Incremented base pointers (advance 32 elements = 64 bytes per K-chunk).
    const __bf16* aH = Hb + (size_t)(rowBase + lm) * DD + 8 * h;   // A rows 0..15
    const __bf16* aL = Lb + (size_t)(rowBase + lm) * DD + 8 * h;
    const __bf16* bH = Hb + (size_t)(colBase + lm) * DD + 8 * h;   // B cols
    const __bf16* bL = Lb + (size_t)(colBase + lm) * DD + 8 * h;

    v8f acc0[4] = {{}, {}, {}, {}};   // M-tile 0 (rows rowBase..+15)
    v8f acc1[4] = {{}, {}, {}, {}};   // M-tile 1 (rows rowBase+16..+31)

    for (int k0 = 0; k0 < DD; k0 += 32) {
        // A fragments: runs at +0 and +16 elems (offset:32 B); tile 1 at
        // +16*DD elems (offset:16384 B). All constant immediate offsets.
        const v16bf Ah0 = cat16(ld8bf(aH), ld8bf(aH + 16));
        const v16bf Al0 = cat16(ld8bf(aL), ld8bf(aL + 16));
        const v16bf Ah1 = cat16(ld8bf(aH + 16 * DD), ld8bf(aH + 16 * DD + 16));
        const v16bf Al1 = cat16(ld8bf(aL + 16 * DD), ld8bf(aL + 16 * DD + 16));
#pragma unroll
        for (int j = 0; j < 4; ++j) {
            // B fragment j: base + j*16*DD elems (immediate), halves at +0/+16.
            const v16bf Bh = cat16(ld8bf(bH + j * 16 * DD),
                                   ld8bf(bH + j * 16 * DD + 16));
            const v16bf Bl = cat16(ld8bf(bL + j * 16 * DD),
                                   ld8bf(bL + j * 16 * DD + 16));
            // G += hi*hi + hi*lo + lo*hi (lo*lo ~ 2^-16 relative; dropped)
            acc0[j] = __builtin_amdgcn_wmma_f32_16x16x32_bf16(
                false, Al0, false, Bh, (short)0, acc0[j], false, false);
            acc0[j] = __builtin_amdgcn_wmma_f32_16x16x32_bf16(
                false, Ah0, false, Bl, (short)0, acc0[j], false, false);
            acc0[j] = __builtin_amdgcn_wmma_f32_16x16x32_bf16(
                false, Ah0, false, Bh, (short)0, acc0[j], false, false);
            acc1[j] = __builtin_amdgcn_wmma_f32_16x16x32_bf16(
                false, Al1, false, Bh, (short)0, acc1[j], false, false);
            acc1[j] = __builtin_amdgcn_wmma_f32_16x16x32_bf16(
                false, Ah1, false, Bl, (short)0, acc1[j], false, false);
            acc1[j] = __builtin_amdgcn_wmma_f32_16x16x32_bf16(
                false, Ah1, false, Bh, (short)0, acc1[j], false, false);
        }
        aH += 32; aL += 32; bH += 32; bL += 32;
    }

    const float* sqb = sq + (size_t)b * NN;
    hyp_epilogue(acc0, sqb, out, b, rowBase,      colBase, h, lm);
    hyp_epilogue(acc1, sqb, out, b, rowBase + 16, colBase, h, lm);
}

// ---------------------------------------------------------------------------
// Fallback GEMM (scratch too small for bf16 arrays): split fp32 in-register.
// ---------------------------------------------------------------------------
__global__ __launch_bounds__(128)
void hyp_dist_f32_kernel(const float* __restrict__ x,
                         const float* __restrict__ sq,
                         float* __restrict__ out) {
    const int lane    = threadIdx.x & 31;
    const int wave    = threadIdx.x >> 5;
    const int b       = blockIdx.z;
    const int rowBase = (blockIdx.y * 4 + wave) * 16;
    const int colBase = blockIdx.x * 64;
    const int h       = lane >> 4;
    const int lm      = lane & 15;

    const float* Xb   = x + (size_t)b * NN * DD;
    const float* arow = Xb + (size_t)(rowBase + lm) * DD;

    v8f acc[4] = {{}, {}, {}, {}};

    for (int k0 = 0; k0 < DD; k0 += 32) {
        float av[16];
        {
            const f4 t0 = ld4(arow + k0 + h * 8);
            const f4 t1 = ld4(arow + k0 + h * 8 + 4);
            const f4 t2 = ld4(arow + k0 + 16 + h * 8);
            const f4 t3 = ld4(arow + k0 + 16 + h * 8 + 4);
#pragma unroll
            for (int e = 0; e < 4; ++e) {
                av[e] = t0[e]; av[4 + e] = t1[e];
                av[8 + e] = t2[e]; av[12 + e] = t3[e];
            }
        }
        v16bf Ahi, Alo;
#pragma unroll
        for (int e = 0; e < 16; ++e) {
            const __bf16 hi = (__bf16)av[e];
            Ahi[e] = hi;
            Alo[e] = (__bf16)(av[e] - (float)hi);
        }
#pragma unroll
        for (int j = 0; j < 4; ++j) {
            const float* bp =
                Xb + (size_t)(colBase + j * 16 + lm) * DD + k0 + h * 8;
            float bv[16];
            {
                const f4 u0 = ld4(bp);
                const f4 u1 = ld4(bp + 4);
                const f4 u2 = ld4(bp + 16);
                const f4 u3 = ld4(bp + 20);
#pragma unroll
                for (int e = 0; e < 4; ++e) {
                    bv[e] = u0[e]; bv[4 + e] = u1[e];
                    bv[8 + e] = u2[e]; bv[12 + e] = u3[e];
                }
            }
            v16bf Bhi, Blo;
#pragma unroll
            for (int e = 0; e < 16; ++e) {
                const __bf16 hi = (__bf16)bv[e];
                Bhi[e] = hi;
                Blo[e] = (__bf16)(bv[e] - (float)hi);
            }
            acc[j] = __builtin_amdgcn_wmma_f32_16x16x32_bf16(
                false, Alo, false, Bhi, (short)0, acc[j], false, false);
            acc[j] = __builtin_amdgcn_wmma_f32_16x16x32_bf16(
                false, Ahi, false, Blo, (short)0, acc[j], false, false);
            acc[j] = __builtin_amdgcn_wmma_f32_16x16x32_bf16(
                false, Ahi, false, Bhi, (short)0, acc[j], false, false);
        }
    }

    hyp_epilogue(acc, sq + (size_t)b * NN, out, b, rowBase, colBase, h, lm);
}

extern "C" void kernel_launch(void* const* d_in, const int* in_sizes, int n_in,
                              void* d_out, int out_size, void* d_ws, size_t ws_size,
                              hipStream_t stream) {
    (void)in_sizes; (void)n_in; (void)out_size;
    const float* x = (const float*)d_in[0];
    float* out     = (float*)d_out;

    const size_t nSq  = (size_t)BB * NN;            // 32768 floats (128 KB)
    const size_t nEl  = (size_t)BB * NN * DD;       // 16.7M elements
    const size_t need = nSq * sizeof(float) + 2 * nEl * sizeof(__bf16); // ~67.2 MB

    float* sq = (float*)d_ws;
    if (ws_size >= need) {
        __bf16* xhi = (__bf16*)((char*)d_ws + nSq * sizeof(float));
        __bf16* xlo = xhi + nEl;
        hyp_prep_kernel<<<dim3((BB * NN) / 8), dim3(256), 0, stream>>>(x, sq, xhi, xlo);
        hyp_dist_bf16_kernel<<<dim3(NN / 64, NN / 128, BB), dim3(128), 0, stream>>>(
            xhi, xlo, sq, out);
    } else {
        hyp_rownorm_kernel<<<dim3((BB * NN) / 8), dim3(256), 0, stream>>>(x, sq);
        hyp_dist_f32_kernel<<<dim3(NN / 64, NN / 64, BB), dim3(128), 0, stream>>>(
            x, sq, out);
    }
}
